// ResGCN_70153995813019
// MI455X (gfx1250) — matrix operand order repaired and checked by reading, not compiled, not run on hardware
//
#include <hip/hip_runtime.h>

typedef _Float16 v16h __attribute__((ext_vector_type(16)));
typedef _Float16 v8h  __attribute__((ext_vector_type(8)));
typedef float    v8f  __attribute__((ext_vector_type(8)));

#define NPOLY 4096
#define PP    128
#define SS    128
#define KF    96      // FEAT=66 padded to 96 (3 k-steps of 32)
#define CCH   64
#define HH    128
#define WW    128
#define NRES  7
#define ROc   4.0f

// ---- WMMA operand loaders (CDNA5 16-bit layouts, CDNA5 ISA 7.12.2) ----
// A 16x32 f16: lane holds A[row][kb..kb+7] and A[row][kb+16..kb+23], kb = kstep*32 + (lane>>4)*8
__device__ __forceinline__ v16h ldA16(const _Float16* p) {
  v8h lo = *(const v8h*)(p);
  v8h hi = *(const v8h*)(p + 16);
  v16h r;
#pragma unroll
  for (int i = 0; i < 8; ++i) { r[i] = lo[i]; r[i + 8] = hi[i]; }
  return r;
}
// B 32x16 f16 (from pre-transposed W^T): lane holds K[kb..kb+7] and K[kb+16..kb+23] for col = lane&15, kb = kstep*32 + (lane>>4)*8
__device__ __forceinline__ v16h ldB16(const _Float16* p) {
  v8h lo = *(const v8h*)(p);
  v8h hi = *(const v8h*)(p + 16);
  v16h r;
#pragma unroll
  for (int i = 0; i < 8; ++i) { r[i] = lo[i]; r[i + 8] = hi[i]; }
  return r;
}
__device__ __forceinline__ v8f wmma16(v16h a, v16h b, v8f c) {
  return __builtin_amdgcn_wmma_f32_16x16x32_f16(false, a, false, b, (short)0, c, false, false);
}

// ---- Prep: convert weights to f16, transposed (N-major) and K-padded ----
// winT  : [4][2][128 n][96 k]    from win_s/win_n [4][66][128]
// wresT : [4][7][2][128 n][128 k] from wres_s/wres_n [4][7][128][128]
__global__ void prep_weights(const float* __restrict__ win_s,
                             const float* __restrict__ win_n,
                             const float* __restrict__ wres_s,
                             const float* __restrict__ wres_n,
                             _Float16* __restrict__ winT,
                             _Float16* __restrict__ wresT) {
  int i = blockIdx.x * blockDim.x + threadIdx.x;
  const int T1 = 4 * 2 * SS * KF;
  const int T2 = 4 * NRES * 2 * SS * SS;
  if (i < T1) {
    int k = i % KF;
    int nn = (i / KF) % SS;
    int w  = (i / (KF * SS)) % 2;
    int m  = i / (KF * SS * 2);
    float v = 0.f;
    if (k < 66) {
      const float* src = w ? win_n : win_s;
      v = src[(m * 66 + k) * SS + nn];
    }
    winT[i] = (_Float16)v;
  } else if (i < T1 + T2) {
    int j = i - T1;
    int k  = j % SS;
    int nn = (j / SS) % SS;
    int w  = (j / (SS * SS)) % 2;
    int l  = (j / (SS * SS * 2)) % NRES;
    int m  = j / (SS * SS * 2 * NRES);
    const float* src = w ? wres_n : wres_s;
    wresT[j] = (_Float16)src[((m * NRES + l) * SS + k) * SS + nn];
  }
}

// ---- One evolution stage: one workgroup (8 wave32) per polygon ----
__global__ __launch_bounds__(256)
void gcn_stage(const float* __restrict__ cnn,      // [16,64,128,128]
               const float* __restrict__ polySrc,  // stage0: i_it_ctrs; else prev pred
               const float* __restrict__ cSrc,     // stage0: c_it_ctrs; else unused
               const int*   __restrict__ ind,
               const _Float16* __restrict__ winT,  // [2][128][96]
               const _Float16* __restrict__ wresT, // [7][2][128][128]
               const float* __restrict__ bin,      // [128]
               const float* __restrict__ bres,     // [7][128]
               const float* __restrict__ wpred,    // [128][2]
               const float* __restrict__ bpred,    // [2]
               float* __restrict__ out,            // [N][128][2]
               int stage)
{
  extern __shared__ char smemRaw[];
  _Float16* sW = (_Float16*)smemRaw;        // 2*128*128 halves (64 KB): staged Ws^T, Wn^T
  _Float16* sA = sW + 2 * SS * SS;          // 128*128 (32 KB): x (stride 96) then h (stride 128)
  _Float16* sM = sA + SS * SS;              // 128*128 (32 KB): aggregated (A·x / A·h)
  float* sPoly = (float*)(sM + SS * SS);    // 256
  float* sC    = sPoly + 2 * PP;            // 256 (c_poly)
  float* sPP   = sC + 2 * PP;               // 512 (x0, y0, wx, wy per point)

  const int tid  = threadIdx.x;
  const int lane = tid & 31;
  const int wv   = tid >> 5;
  const int n    = blockIdx.x;
  const float inv9 = 1.0f / 9.0f;

  // ---- stage poly / c_poly ----
  {
    float v = polySrc[n * (PP * 2) + tid];
    if (stage != 0) v *= (1.0f / ROc);
    sPoly[tid] = v;
    if (stage == 0) sC[tid] = cSrc[n * (PP * 2) + tid];
  }
  __syncthreads();
  if (stage != 0) {   // uniform branch
    if (tid < 2) {
      float mn = sPoly[tid];
      for (int p = 1; p < PP; ++p) mn = fminf(mn, sPoly[p * 2 + tid]);
      sPP[tid] = mn;
    }
    __syncthreads();
    sC[tid] = sPoly[tid] - sPP[tid & 1];
    __syncthreads();
  }
  // per-point bilinear params (px = x - 0.5, py = y - 0.5)
  if (tid < PP) {
    float px = sPoly[tid * 2 + 0] - 0.5f;
    float py = sPoly[tid * 2 + 1] - 0.5f;
    float fx = floorf(px), fy = floorf(py);
    sPP[0 * PP + tid] = fx;
    sPP[1 * PP + tid] = fy;
    sPP[2 * PP + tid] = px - fx;
    sPP[3 * PP + tid] = py - fy;
  }
  // stage input-layer weights while params settle
  for (int i = tid; i < 2 * SS * KF; i += 256) sW[i] = winT[i];
  __syncthreads();

  // ---- build x (128 x 96, f16, zero-padded) into sA ----
  const float* fbase = cnn + (size_t)ind[n] * CCH * HH * WW;
  for (int i = tid; i < PP * KF; i += 256) {
    int p = i / KF;
    int c = i - p * KF;
    float val;
    if (c < CCH) {
      float fx = sPP[0 * PP + p], fy = sPP[1 * PP + p];
      float wx = sPP[2 * PP + p], wy = sPP[3 * PP + p];
      int x0 = (int)fx, y0 = (int)fy;
      int x0i = min(max(x0, 0), WW - 1);
      int x1i = min(max(x0 + 1, 0), WW - 1);
      int y0i = min(max(y0, 0), HH - 1);
      int y1i = min(max(y0 + 1, 0), HH - 1);
      const float* ch = fbase + (size_t)c * HH * WW;
      float f00 = ch[y0i * WW + x0i];
      float f01 = ch[y0i * WW + x1i];
      float f10 = ch[y1i * WW + x0i];
      float f11 = ch[y1i * WW + x1i];
      val = f00 * (1.f - wx) * (1.f - wy) + f01 * wx * (1.f - wy)
          + f10 * (1.f - wx) * wy + f11 * wx * wy;
    } else if (c < CCH + 2) {
      val = sC[p * 2 + (c - CCH)] * ROc;
    } else {
      val = 0.f;
    }
    sA[p * KF + c] = (_Float16)val;
  }
  __syncthreads();

  // ---- mx = ring-average(x)  (A is 9-tap circulant / 9) ----
  for (int i = tid; i < PP * KF; i += 256) {
    int p = i / KF;
    int c = i - p * KF;
    float s = 0.f;
#pragma unroll
    for (int d = -4; d <= 4; ++d) s += (float)sA[((p + d) & (PP - 1)) * KF + c];
    sM[p * KF + c] = (_Float16)(s * inv9);
  }
  __syncthreads();

  const int rsel = lane & 15;   // A row-in-tile == B col-in-tile
  const int hsel = lane >> 4;

  // ---- input layer: h = relu(x@Ws_in + mx@Wn_in + b_in) ----
  {
    v8f acc[8];
#pragma unroll
    for (int j = 0; j < 8; ++j) { v8f z = {}; acc[j] = z; }
    const int arow = wv * 16 + rsel;
#pragma unroll
    for (int k = 0; k < 3; ++k) {
      int kbA = k * 32 + hsel * 8;
      v16h aX = ldA16(&sA[arow * KF + kbA]);
      v16h aM = ldA16(&sM[arow * KF + kbA]);
      int kbB = k * 32 + hsel * 8;
#pragma unroll
      for (int j = 0; j < 8; ++j) {
        int ncol = j * 16 + rsel;
        v16h bS = ldB16(&sW[ncol * KF + kbB]);
        v16h bN = ldB16(&sW[(SS + ncol) * KF + kbB]);
        acc[j] = wmma16(aX, bS, acc[j]);
        acc[j] = wmma16(aM, bN, acc[j]);
      }
    }
    __syncthreads();  // all waves done reading stride-96 x before stride-128 h store
#pragma unroll
    for (int j = 0; j < 8; ++j) {
      int ncol = j * 16 + rsel;
      float b = bin[ncol];
#pragma unroll
      for (int v = 0; v < 8; ++v) {
        int row = wv * 16 + v + 8 * hsel;
        float h = acc[j][v] + b;
        sA[row * SS + ncol] = (_Float16)(h > 0.f ? h : 0.f);
      }
    }
  }
  __syncthreads();

  // ---- 7 residual layers ----
  for (int l = 0; l < NRES; ++l) {
    // stage this layer's Ws^T, Wn^T into LDS (vectorized b128 copies)
    {
      const v8h* src = (const v8h*)(wresT + (size_t)l * 2 * SS * SS);
      v8h* dst = (v8h*)sW;
      for (int i = tid; i < (2 * SS * SS) / 8; i += 256) dst[i] = src[i];
    }
    // m = ring-average(h), running 9-wide window per column
    {
      int col = tid & (SS - 1);
      int r0 = (tid >> 7) * 64;
      float s = 0.f;
#pragma unroll
      for (int d = -4; d <= 4; ++d) s += (float)sA[((r0 + d) & (PP - 1)) * SS + col];
      for (int r = r0; r < r0 + 64; ++r) {
        sM[r * SS + col] = (_Float16)(s * inv9);
        s += (float)sA[((r + 5) & (PP - 1)) * SS + col];
        s -= (float)sA[((r - 4) & (PP - 1)) * SS + col];
      }
    }
    __syncthreads();
    // hnew = relu(h@Ws + m@Wn + b) + h
    v8f acc[8];
#pragma unroll
    for (int j = 0; j < 8; ++j) { v8f z = {}; acc[j] = z; }
    const int arow = wv * 16 + rsel;
#pragma unroll
    for (int k = 0; k < 4; ++k) {
      int kbA = k * 32 + hsel * 8;
      v16h aH = ldA16(&sA[arow * SS + kbA]);
      v16h aM = ldA16(&sM[arow * SS + kbA]);
      int kbB = k * 32 + hsel * 8;
#pragma unroll
      for (int j = 0; j < 8; ++j) {
        int ncol = j * 16 + rsel;
        v16h bS = ldB16(&sW[ncol * SS + kbB]);
        v16h bN = ldB16(&sW[(SS + ncol) * SS + kbB]);
        acc[j] = wmma16(aH, bS, acc[j]);
        acc[j] = wmma16(aM, bN, acc[j]);
      }
    }
    __syncthreads();
    const float* bl = bres + l * SS;
#pragma unroll
    for (int j = 0; j < 8; ++j) {
      int ncol = j * 16 + rsel;
      float b = bl[ncol];
#pragma unroll
      for (int v = 0; v < 8; ++v) {
        int row = wv * 16 + v + 8 * hsel;
        float hold = (float)sA[row * SS + ncol];
        float h = acc[j][v] + b;
        sA[row * SS + ncol] = (_Float16)((h > 0.f ? h : 0.f) + hold);
      }
    }
    __syncthreads();
  }

  // ---- prediction head + pred = poly*RO + off ----
  {
    int p = tid >> 1, a = tid & 1;
    float acc = bpred[a];
    for (int s2 = 0; s2 < SS; ++s2)
      acc += (float)sA[p * SS + s2] * wpred[s2 * 2 + a];
    out[n * (PP * 2) + tid] = sPoly[tid] * ROc + acc;
  }
}

extern "C" void kernel_launch(void* const* d_in, const int* in_sizes, int n_in,
                              void* d_out, int out_size, void* d_ws, size_t ws_size,
                              hipStream_t stream) {
  const float* cnn    = (const float*)d_in[0];
  const float* i_it   = (const float*)d_in[1];
  const float* c_it   = (const float*)d_in[2];
  const int*   ind    = (const int*)d_in[3];
  const float* win_s  = (const float*)d_in[4];
  const float* win_n  = (const float*)d_in[5];
  const float* bin_   = (const float*)d_in[6];
  const float* wres_s = (const float*)d_in[7];
  const float* wres_n = (const float*)d_in[8];
  const float* bres   = (const float*)d_in[9];
  const float* wpred  = (const float*)d_in[10];
  const float* bpred  = (const float*)d_in[11];
  float* out = (float*)d_out;

  _Float16* winT  = (_Float16*)d_ws;
  _Float16* wresT = winT + 4 * 2 * SS * KF;

  {
    int total  = 4 * 2 * SS * KF + 4 * NRES * 2 * SS * SS;
    int blocks = (total + 255) / 256;
    prep_weights<<<blocks, 256, 0, stream>>>(win_s, win_n, wres_s, wres_n, winT, wresT);
  }

  const size_t smemBytes = (size_t)(2 * SS * SS + SS * SS + SS * SS) * sizeof(_Float16)
                         + (size_t)(2 * PP + 2 * PP + 4 * PP) * sizeof(float);

  for (int m = 0; m < 4; ++m) {
    const float* polySrc = (m == 0) ? i_it : (out + (size_t)(m - 1) * NPOLY * PP * 2);
    const float* cSrc    = (m == 0) ? c_it : i_it;  // unused for m>0
    gcn_stage<<<NPOLY, 256, smemBytes, stream>>>(
        cnn, polySrc, cSrc, ind,
        winT + (size_t)m * 2 * SS * KF,
        wresT + (size_t)m * NRES * 2 * SS * SS,
        bin_ + m * SS,
        bres + m * NRES * SS,
        wpred + m * SS * 2,
        bpred + m * 2,
        out + (size_t)m * NPOLY * PP * 2,
        m);
  }
}
